// FraudGNN_63625645523668
// MI455X (gfx1250) — matrix operand order repaired and checked by reading, not compiled, not run on hardware
//
#include <hip/hip_runtime.h>

// ---------------------------------------------------------------------------
// FraudGNN (3-layer GraphSAGE, mean aggregation) for MI455X / gfx1250.
//
// v2: LDS-staged A tiles. All 8 waves of a GEMM block share the same 16
// activation rows, so the 16x128 h-tile and agg-tile are converted to bf16
// once per block into LDS (8 KB of the WGP's 320 KB), then each wave builds
// its WMMA A-fragments with 16B ds reads. This removes the 8x-redundant
// global loads + f32->bf16 conversions seen in v1's assembly.
// Matrix math: V_WMMA_F32_16X16X32_BF16 (bf16 in, f32 accumulate).
// ---------------------------------------------------------------------------

typedef __attribute__((ext_vector_type(16))) __bf16 v16bf;
typedef __attribute__((ext_vector_type(8)))  __bf16 v8bf;
typedef __attribute__((ext_vector_type(8)))  float  v8f;

#define CH 128   // feature width (IN_CH == HID == 128)

// ---------------------------------------------------------------- utilities
__global__ void k_zero_f32(float* __restrict__ p, long n) {
  long i = (long)blockIdx.x * blockDim.x + threadIdx.x;
  long stride = (long)gridDim.x * blockDim.x;
  for (; i < n; i += stride) p[i] = 0.0f;
}

// deg[dst] += 1 over edges
__global__ void k_deg(const int* __restrict__ dst, float* __restrict__ deg, int E) {
  int e = blockIdx.x * blockDim.x + threadIdx.x;
  if (e < E) atomicAdd(&deg[dst[e]], 1.0f);
}

// deg -> inv_deg in place
__global__ void k_invdeg(float* __restrict__ deg, int n) {
  int i = blockIdx.x * blockDim.x + threadIdx.x;
  if (i < n) { float d = deg[i]; deg[i] = d > 0.0f ? 1.0f / d : 0.0f; }
}

// W (K x N, f32, row-major) -> Wt (N x K, bf16) so that each B-fragment lane
// load is one contiguous 32B v16bf load.
__global__ void k_wt_bf16(const float* __restrict__ W, __bf16* __restrict__ Wt) {
  int t = blockIdx.x * blockDim.x + threadIdx.x;   // 128*128 threads
  int k = t >> 7, n = t & 127;
  Wt[n * CH + k] = (__bf16)W[k * CH + n];
}

// ------------------------------------------------------------- edge scatter
// One wave (32 lanes) per edge. src/dst are wave-uniform: force them scalar
// (readfirstlane) so row-base math lives in SGPRs and the gather uses the
// SGPR-base + VGPR-offset addressing form. Each lane moves 4 channels.
__global__ void k_scatter(const float* __restrict__ h, const int* __restrict__ src,
                          const int* __restrict__ dst, float* __restrict__ msg, int E) {
  int e = blockIdx.x * 8 + (threadIdx.x >> 5);
  if (e >= E) return;
  int lane = threadIdx.x & 31;
  int s = __builtin_amdgcn_readfirstlane(src[e]);
  int d = __builtin_amdgcn_readfirstlane(dst[e]);
  float4 v = *(const float4*)(h + (long)s * CH + lane * 4);
  float* o = msg + (long)d * CH + lane * 4;
  atomicAdd(o + 0, v.x);
  atomicAdd(o + 1, v.y);
  atomicAdd(o + 2, v.z);
  atomicAdd(o + 3, v.w);
}

// ------------------------------------------------------- fused SAGE + WMMA
// A-fragment from an LDS bf16 tile (contiguous k order, matching the B loads):
//   lanes 0-15  : row M=lane,    K = kb+0..7  (v0..v3) and kb+8..15 (v4..v7)
//   lanes 16-31 : row M=lane-16, K = kb+16..23 (v0..v3) and kb+24..31 (v4..v7)
// => two 16B ds reads per fragment, glued with a shufflevector.
__device__ __forceinline__ v16bf a_frag_lds(const __bf16* tile, int lan, int k0) {
  v8bf lo = *(const v8bf*)(tile + lan * CH + k0);
  v8bf hi = *(const v8bf*)(tile + lan * CH + k0 + 8);
  return __builtin_shufflevector(lo, hi, 0, 1, 2, 3, 4, 5, 6, 7,
                                 8, 9, 10, 11, 12, 13, 14, 15);
}

// hout = relu( (msg*inv_deg) @ Wl + bl + h @ Wr )
// Block: 256 threads = 8 waves covering a 16x128 output tile; each wave owns
// one 16x16 sub-tile. K-loop: 4 steps of 32, 2 WMMAs per step.
__global__ __launch_bounds__(256) void k_sage_gemm(
    const float* __restrict__ h, const float* __restrict__ msg,
    const float* __restrict__ invdeg,
    const __bf16* __restrict__ Wlt, const __bf16* __restrict__ Wrt,
    const float* __restrict__ bl, float* __restrict__ hout, int N) {
  __shared__ __bf16 sh[16 * CH];   // bf16 h tile       (4 KB)
  __shared__ __bf16 sm[16 * CH];   // bf16 agg tile     (4 KB)

  int m0 = blockIdx.x * 16;

  // ---- cooperative stage: 256 threads x 8 elements per tile ----
  {
    int t    = threadIdx.x;
    int row  = t >> 4;           // 0..15
    int col0 = (t & 15) * 8;     // 8 channels per thread
    int gr   = m0 + row;
    if (gr >= N) gr = N - 1;     // clamp (N=50000 is divisible by 16)
    float sc = invdeg[gr];
    float4 h0 = *(const float4*)(h   + (long)gr * CH + col0);
    float4 h1 = *(const float4*)(h   + (long)gr * CH + col0 + 4);
    float4 a0 = *(const float4*)(msg + (long)gr * CH + col0);
    float4 a1 = *(const float4*)(msg + (long)gr * CH + col0 + 4);
    v8bf ph, pa;
    ph[0] = (__bf16)h0.x; ph[1] = (__bf16)h0.y; ph[2] = (__bf16)h0.z; ph[3] = (__bf16)h0.w;
    ph[4] = (__bf16)h1.x; ph[5] = (__bf16)h1.y; ph[6] = (__bf16)h1.z; ph[7] = (__bf16)h1.w;
    pa[0] = (__bf16)(a0.x * sc); pa[1] = (__bf16)(a0.y * sc);
    pa[2] = (__bf16)(a0.z * sc); pa[3] = (__bf16)(a0.w * sc);
    pa[4] = (__bf16)(a1.x * sc); pa[5] = (__bf16)(a1.y * sc);
    pa[6] = (__bf16)(a1.z * sc); pa[7] = (__bf16)(a1.w * sc);
    *(v8bf*)(sh + row * CH + col0) = ph;
    *(v8bf*)(sm + row * CH + col0) = pa;
  }
  __syncthreads();

  // ---- per-wave WMMA ----
  int wave = threadIdx.x >> 5;
  int lane = threadIdx.x & 31;
  int hi   = lane >> 4;        // half-wave selector
  int lan  = lane & 15;
  int ncol = wave * 16 + lan;  // this lane's output column (B/C/D N index)

  float bv = bl[ncol];                       // bias broadcast into the C tile
  v8f acc = {bv, bv, bv, bv, bv, bv, bv, bv};

#pragma unroll
  for (int kb = 0; kb < CH; kb += 32) {
    int k0 = kb + (hi ? 16 : 0);
    v16bf a_h   = a_frag_lds(sh, lan, k0);
    v16bf a_agg = a_frag_lds(sm, lan, k0);
    // B layout (32x16 KxN, bf16): lane n<16 -> N=n, K=kb..kb+15;
    // lane n+16 -> N=n, K=kb+16..kb+31. With W^T[n][k]: one 32B load.
    int kstart = kb + (hi ? 16 : 0);
    v16bf b_l = *(const v16bf*)(Wlt + (long)ncol * CH + kstart);
    v16bf b_r = *(const v16bf*)(Wrt + (long)ncol * CH + kstart);
    acc = __builtin_amdgcn_wmma_f32_16x16x32_bf16(false, a_agg, false, b_l,
                                                  (short)0, acc, false, false);
    acc = __builtin_amdgcn_wmma_f32_16x16x32_bf16(false, a_h,   false, b_r,
                                                  (short)0, acc, false, false);
  }

  // C/D layout: lane<16 -> (M=r, N=lan); lane>=16 -> (M=r+8, N=lan)
#pragma unroll
  for (int r = 0; r < 8; ++r) {
    int row = m0 + r + hi * 8;
    if (row < N) {
      float v = acc[r];
      hout[(long)row * CH + ncol] = v > 0.0f ? v : 0.0f;
    }
  }
}

// ------------------------------------------------------------------- head
// out[node] = dot(h[node], Wh) + bh  — one wave per node, float4 + shuffle.
__global__ void k_head(const float* __restrict__ h, const float* __restrict__ Wh,
                       const float* __restrict__ bh, float* __restrict__ out, int N) {
  int wave = threadIdx.x >> 5, lane = threadIdx.x & 31;
  int node = blockIdx.x * 8 + wave;
  if (node >= N) return;
  float4 hv = *(const float4*)(h + (long)node * CH + lane * 4);
  float4 wv = *(const float4*)(Wh + lane * 4);
  float v = hv.x * wv.x + hv.y * wv.y + hv.z * wv.z + hv.w * wv.w;
#pragma unroll
  for (int off = 16; off; off >>= 1) v += __shfl_xor(v, off, 32);
  if (lane == 0) out[node] = v + bh[0];
}

// ------------------------------------------------------------------ launch
extern "C" void kernel_launch(void* const* d_in, const int* in_sizes, int n_in,
                              void* d_out, int out_size, void* d_ws, size_t ws_size,
                              hipStream_t stream) {
  const float* x    = (const float*)d_in[0];
  const int*   edge = (const int*)d_in[1];
  const float* Wl[3] = {(const float*)d_in[2], (const float*)d_in[5], (const float*)d_in[8]};
  const float* Wr[3] = {(const float*)d_in[3], (const float*)d_in[6], (const float*)d_in[9]};
  const float* bl[3] = {(const float*)d_in[4], (const float*)d_in[7], (const float*)d_in[10]};
  const float* Wh = (const float*)d_in[11];
  const float* bh = (const float*)d_in[12];

  const int N = in_sizes[0] / CH;   // 50000
  const int E = in_sizes[1] / 2;    // 800000
  const int* src = edge;
  const int* dst = edge + E;

  // ---- workspace carve-out (256B aligned slabs) ----
  char* ws = (char*)d_ws;
  size_t off = 0;
  auto carve = [&](size_t bytes) -> void* {
    void* p = ws + off;
    off = (off + bytes + 255) & ~(size_t)255;
    return p;
  };
  float*  msg = (float*)carve((size_t)N * CH * sizeof(float));   // 25.6 MB
  float*  hA  = (float*)carve((size_t)N * CH * sizeof(float));   // 25.6 MB
  float*  hB  = (float*)carve((size_t)N * CH * sizeof(float));   // 25.6 MB
  float*  deg = (float*)carve((size_t)N * sizeof(float));        // 0.2 MB
  __bf16* Wt  = (__bf16*)carve((size_t)6 * CH * CH * sizeof(__bf16)); // 0.2 MB
  __bf16* Wlt[3];
  __bf16* Wrt[3];
  for (int l = 0; l < 3; ++l) {
    Wlt[l] = Wt + (size_t)(2 * l + 0) * CH * CH;
    Wrt[l] = Wt + (size_t)(2 * l + 1) * CH * CH;
  }

  // ---- one-time prep: bf16 transposed weights, inv_deg ----
  for (int l = 0; l < 3; ++l) {
    k_wt_bf16<<<CH * CH / 256, 256, 0, stream>>>(Wl[l], Wlt[l]);
    k_wt_bf16<<<CH * CH / 256, 256, 0, stream>>>(Wr[l], Wrt[l]);
  }
  k_zero_f32<<<256, 256, 0, stream>>>(deg, N);
  k_deg<<<(E + 255) / 256, 256, 0, stream>>>(dst, deg, E);
  k_invdeg<<<(N + 255) / 256, 256, 0, stream>>>(deg, N);

  // ---- 3 SAGE layers, ping-pong h between hA/hB ----
  const float* hcur = x;
  float* hnext = hA;
  for (int l = 0; l < 3; ++l) {
    k_zero_f32<<<1024, 256, 0, stream>>>(msg, (long)N * CH);
    k_scatter<<<(E + 7) / 8, 256, 0, stream>>>(hcur, src, dst, msg, E);
    k_sage_gemm<<<(N + 15) / 16, 256, 0, stream>>>(hcur, msg, deg, Wlt[l], Wrt[l],
                                                   bl[l], hnext, N);
    hcur = hnext;
    hnext = (hnext == hA) ? hB : hA;
  }

  // ---- head ----
  k_head<<<(N + 7) / 8, 256, 0, stream>>>(hcur, Wh, bh, (float*)d_out, N);
}
